// FeaturesStructuringLayer_70712341561604
// MI455X (gfx1250) — matrix operand order repaired and checked by reading, not compiled, not run on hardware
//
#include <hip/hip_runtime.h>
#include <hip/hip_bf16.h>

#define BATCH 16
#define MPTS  32768
#define KPT   85
#define NF    64
#define DOUT  16

typedef __attribute__((ext_vector_type(16))) __bf16 v16bf;
typedef __attribute__((ext_vector_type(8)))  float  v8f;

// ---------------------------------------------------------------- zero ws
__global__ void fsl_zero_kernel(float* __restrict__ p, int n) {
    int i = blockIdx.x * 256 + threadIdx.x;
    if (i < n) p[i] = 0.0f;
}

// ---------------------------------------------------------------- pass 1
// QFacc[b,k,f] += sum_m QD[b,m,k] * (F[b,m,f]*omegaF[f])
// grid (32, BATCH), block 256 (8 waves): waves 0-5 own 16 k-rows each (K
// padded to 96) and do the WMMAs; all 8 waves do the (bandwidth-bound)
// staging. M-tile = 128 points per iteration (four 32-wide contraction steps).
__global__ __launch_bounds__(256) void fsl_pass1_kernel(
    const float* __restrict__ X, const float* __restrict__ F,
    const float* __restrict__ QX, const float* __restrict__ omegaD,
    const float* __restrict__ omegaF, float* __restrict__ QFacc)
{
    __shared__ float sQX0[96], sQX1[96], sQX2[96], sInv[96];
    __shared__ float sOmF[NF];
    __shared__ __align__(16) float sX[128 * 3];
    __shared__ __align__(16) __bf16 sFT[NF * 136];  // [f][m], m padded 128->136 (16B rows)

    const int tid = threadIdx.x;
    const int b = blockIdx.y;

    if (tid < 96) {
        if (tid < KPT) {
            sQX0[tid] = QX[tid * 3 + 0];
            sQX1[tid] = QX[tid * 3 + 1];
            sQX2[tid] = QX[tid * 3 + 2];
            float w = omegaD[tid];
            sInv[tid] = 1.0f / (w * w);
        } else {
            sQX0[tid] = 0.0f; sQX1[tid] = 0.0f; sQX2[tid] = 0.0f; sInv[tid] = 0.0f;
        }
    }
    if (tid < NF) sOmF[tid] = omegaF[tid];
    __syncthreads();

    const int lane = tid & 31;
    const int wv   = tid >> 5;                 // 0..7; waves 0..5 compute
    const int hp   = lane >> 4;                // half-wave select
    const int l16  = lane & 15;
    const int krow = (wv < 6 ? wv : 5) * 16 + l16;   // clamped for non-compute waves
    const float qx0 = sQX0[krow], qx1 = sQX1[krow], qx2 = sQX2[krow];
    const float inv = sInv[krow];
    const bool  kvalid = (wv < 6) && (krow < KPT);

    v8f c[4] = {};

    const size_t bm = (size_t)b * MPTS + (size_t)blockIdx.x * 1024;
    for (int it = 0; it < 8; ++it) {
        const size_t m0 = bm + it * 128;
        __syncthreads();
        // stage X tile: 128 points * 3 floats = 96 float4
        if (tid < 96) {
            float4 v = *(const float4*)&X[m0 * 3 + tid * 4];
            *(float4*)&sX[tid * 4] = v;
        }
        // prefetch next iteration's F tile: 32KB = 256 x 128B lines, one/thread
        if (it < 7)
            __builtin_prefetch((const char*)(F + (m0 + 128) * NF) + tid * 128, 0, 3);
        // stage F tile: 8 independent float4 loads first, then convert+store
        float4 fv[8];
        #pragma unroll
        for (int kk = 0; kk < 8; ++kk) {
            int idx = tid + kk * 256;
            fv[kk] = *(const float4*)&F[(m0 + (idx >> 4)) * NF + (idx & 15) * 4];
        }
        __builtin_amdgcn_sched_barrier(0);   // keep all 8 loads in flight
        #pragma unroll
        for (int kk = 0; kk < 8; ++kk) {
            int idx = tid + kk * 256;
            int ml = idx >> 4, f = (idx & 15) * 4;
            sFT[(f + 0) * 136 + ml] = (__bf16)(fv[kk].x * sOmF[f + 0]);
            sFT[(f + 1) * 136 + ml] = (__bf16)(fv[kk].y * sOmF[f + 1]);
            sFT[(f + 2) * 136 + ml] = (__bf16)(fv[kk].z * sOmF[f + 2]);
            sFT[(f + 3) * 136 + ml] = (__bf16)(fv[kk].w * sOmF[f + 3]);
        }
        __syncthreads();

        if (wv < 6) {
            #pragma unroll
            for (int sub = 0; sub < 4; ++sub) {
                // issue all 4 B-fragment loads first; LDS latency hides under exp chain
                union { uint4 q[2]; v16bf v; } u[4];
                #pragma unroll
                for (int t = 0; t < 4; ++t) {
                    const uint4* p =
                        (const uint4*)(&sFT[(t * 16 + l16) * 136 + sub * 32 + hp * 16]);
                    u[t].q[0] = p[0]; u[t].q[1] = p[1];
                }
                __builtin_amdgcn_sched_barrier(0);
                // A fragment: 16x32 bf16, rows = k, contraction = local m
                // layout: lanes<16 -> K 0-15 ; lanes>=16 -> K 16-31 (matches B)
                v16bf a = {};
                #pragma unroll
                for (int i = 0; i < 16; ++i) {
                    int ml = sub * 32 + hp * 16 + i;
                    float dx = sX[ml * 3 + 0] - qx0;
                    float dy = sX[ml * 3 + 1] - qx1;
                    float dz = sX[ml * 3 + 2] - qx2;
                    float d2 = dx * dx + dy * dy + dz * dz;
                    float qd = kvalid ? __expf(-d2 * inv) : 0.0f;
                    a[i] = (__bf16)qd;
                }
                #pragma unroll
                for (int t = 0; t < 4; ++t)
                    c[t] = __builtin_amdgcn_wmma_f32_16x16x32_bf16(
                               false, a, false, u[t].v, (short)0, c[t], false, false);
            }
        }
    }

    // epilogue: atomic-accumulate f32 C tiles into global QF
    if (wv < 6) {
        #pragma unroll
        for (int t = 0; t < 4; ++t) {
            #pragma unroll
            for (int j = 0; j < 8; ++j) {
                int row = wv * 16 + hp * 8 + j;   // C/D layout: row = half*8 + vgpr
                if (row < KPT) {
                    int col = t * 16 + l16;
                    unsafeAtomicAdd(&QFacc[((size_t)b * KPT + row) * NF + col], c[t][j]);
                }
            }
        }
    }
}

// ---------------------------------------------------------------- QFW = QF @ QW
__global__ __launch_bounds__(256) void fsl_qfw_kernel(
    const float* __restrict__ QFacc, const float* __restrict__ QW,
    float* __restrict__ QFW)
{
    int b = blockIdx.x;
    for (int idx = threadIdx.x; idx < KPT * DOUT; idx += 256) {
        int k = idx >> 4, d = idx & 15;
        const float* qf = &QFacc[((size_t)b * KPT + k) * NF];
        float s = 0.0f;
        #pragma unroll
        for (int f = 0; f < NF; ++f) s += qf[f] * QW[f * DOUT + d];
        QFW[(size_t)b * KPT * DOUT + idx] = s;
    }
}

// ---------------------------------------------------------------- pass 2
// QY[m,:] = QD[m,:] @ QFW ; out = [F | QY]
// grid (MPTS/128, BATCH), block 256 (8 waves); each wave makes one 16x16 QY tile.
__global__ __launch_bounds__(256) void fsl_pass2_kernel(
    const float* __restrict__ X, const float* __restrict__ F,
    const float* __restrict__ QX, const float* __restrict__ omegaD,
    const float* __restrict__ QFW, float* __restrict__ out)
{
    __shared__ float sQX0[96], sQX1[96], sQX2[96], sInv[96];
    __shared__ __align__(16) __bf16 sQT[16 * 104];  // [d][k], k padded 96->104
    __shared__ __align__(16) float sXP[128 * 3];
    __shared__ __align__(16) float sQY[128 * 16];

    const int tid = threadIdx.x;
    const int b = blockIdx.y;
    const int mBase = blockIdx.x * 128;
    const size_t rowBase = (size_t)b * MPTS + mBase;

    // prefetch this block's 32KB F slab (128 rows x 256B)
    __builtin_prefetch((const char*)(F + rowBase * NF) + tid * 128, 0, 3);

    if (tid < 96) {
        if (tid < KPT) {
            sQX0[tid] = QX[tid * 3 + 0];
            sQX1[tid] = QX[tid * 3 + 1];
            sQX2[tid] = QX[tid * 3 + 2];
            float w = omegaD[tid];
            sInv[tid] = 1.0f / (w * w);
        } else {
            sQX0[tid] = 0.0f; sQX1[tid] = 0.0f; sQX2[tid] = 0.0f; sInv[tid] = 0.0f;
        }
    }
    // stage X for the block's 128 points: 384 floats = 96 float4 loads
    if (tid < 96) {
        float4 v = *(const float4*)&X[rowBase * 3 + tid * 4];
        *(float4*)&sXP[tid * 4] = v;
    }
    // stage QFW^T as bf16 [d][k]
    for (int idx = tid; idx < 16 * 96; idx += 256) {
        int d = idx / 96, k = idx % 96;
        float v = (k < KPT) ? QFW[((size_t)b * KPT + k) * DOUT + d] : 0.0f;
        sQT[d * 104 + k] = (__bf16)v;
    }
    __syncthreads();

    const int lane = tid & 31;
    const int wv   = tid >> 5;
    const int hp   = lane >> 4;
    const int l16  = lane & 15;

    const int pl = wv * 16 + l16;                    // this lane's local point (A row)
    const float x0 = sXP[pl * 3 + 0], x1 = sXP[pl * 3 + 1], x2 = sXP[pl * 3 + 2];

    v8f c = {};
    #pragma unroll
    for (int ch = 0; ch < 3; ++ch) {                 // K = 96 = 3 x 32
        // B fragment first, so LDS latency hides under the exp chain
        const uint4* p = (const uint4*)(&sQT[l16 * 104 + ch * 32 + hp * 16]); // 16B aligned
        union { uint4 q[2]; v16bf v; } u;
        u.q[0] = p[0]; u.q[1] = p[1];
        __builtin_amdgcn_sched_barrier(0);
        v16bf a = {};
        #pragma unroll
        for (int i = 0; i < 16; ++i) {
            int k = ch * 32 + hp * 16 + i;
            float dx = x0 - sQX0[k];
            float dy = x1 - sQX1[k];
            float dz = x2 - sQX2[k];
            float d2 = dx * dx + dy * dy + dz * dz;
            float qd = (k < KPT) ? __expf(-d2 * sInv[k]) : 0.0f;
            a[i] = (__bf16)qd;
        }
        c = __builtin_amdgcn_wmma_f32_16x16x32_bf16(
                false, a, false, u.v, (short)0, c, false, false);
    }

    // stage QY tiles to LDS so output rows can be written fully coalesced
    #pragma unroll
    for (int j = 0; j < 8; ++j) {
        int prow = wv * 16 + hp * 8 + j;             // C/D layout: row = half*8 + vgpr
        sQY[prow * 16 + l16] = c[j];
    }
    __syncthreads();

    // copy F into out[:, 0:64]: 2048 float4, 8 per thread, unrolled
    #pragma unroll
    for (int kk = 0; kk < 8; ++kk) {
        int idx = tid + kk * 256;
        int pnt = idx >> 4, q = idx & 15;
        float4 v = *(const float4*)&F[(rowBase + pnt) * NF + q * 4];
        *(float4*)&out[(rowBase + pnt) * (NF + DOUT) + q * 4] = v;
    }
    // write QY into out[:, 64:80]: 512 float4, 2 per thread
    #pragma unroll
    for (int kk = 0; kk < 2; ++kk) {
        int idx = tid + kk * 256;
        int pnt = idx >> 2, q = idx & 3;
        float4 v = *(const float4*)&sQY[pnt * 16 + q * 4];
        *(float4*)&out[(rowBase + pnt) * (NF + DOUT) + NF + q * 4] = v;
    }
}

// ---------------------------------------------------------------- launch
extern "C" void kernel_launch(void* const* d_in, const int* in_sizes, int n_in,
                              void* d_out, int out_size, void* d_ws, size_t ws_size,
                              hipStream_t stream) {
    const float* X      = (const float*)d_in[0];
    const float* F      = (const float*)d_in[1];
    const float* QX     = (const float*)d_in[2];
    const float* omegaD = (const float*)d_in[3];
    const float* omegaF = (const float*)d_in[4];
    const float* QW     = (const float*)d_in[5];
    float* out = (float*)d_out;

    float* QFacc = (float*)d_ws;                      // [B][85][64] f32
    float* QFWp  = QFacc + (size_t)BATCH * KPT * NF;  // [B][85][16] f32

    const int nz = BATCH * KPT * NF;
    fsl_zero_kernel<<<(nz + 255) / 256, 256, 0, stream>>>(QFacc, nz);
    fsl_pass1_kernel<<<dim3(32, BATCH), 256, 0, stream>>>(X, F, QX, omegaD, omegaF, QFacc);
    fsl_qfw_kernel<<<BATCH, 256, 0, stream>>>(QFacc, QW, QFWp);
    fsl_pass2_kernel<<<dim3(MPTS / 128, BATCH), 256, 0, stream>>>(X, F, QX, omegaD, QFWp, out);
}
